// SrvfClasNet_15882789061307
// MI455X (gfx1250) — matrix operand order repaired and checked by reading, not compiled, not run on hardware
//
#include <hip/hip_runtime.h>
#include <hip/hip_bf16.h>
#include <math.h>

// ---------------------------------------------------------------------------
// SRVF classification net forward pass for MI455X (gfx1250, wave32, WMMA).
// conv2 / conv3 / FC are WMMA f16 GEMMs with f32 accumulation; BatchNorm is
// two-phase (stats via LDS+global atomics, then folded scale/shift applied on
// the fly when the consumer reads the f16 activations).
// Epilogues store packed v8h (16B) per lane: D-fragment rows are contiguous.
// ---------------------------------------------------------------------------

#define BATCH 16384
#define SEQL  150
#define EPSBN 1e-5f

typedef _Float16 h16;
typedef __attribute__((ext_vector_type(16))) _Float16 v16h;
typedef __attribute__((ext_vector_type(8)))  _Float16 v8h;
typedef __attribute__((ext_vector_type(8)))  float    v8f;

// ---- WMMA fragment loaders (ISA 7.12.2 layouts, wave32) -------------------
// A 16x32 f16, row-major source with leading dim `ld` (halves):
//   lanes 0-15 : row M=lane,     halves = K {0..7, 16..23}
//   lanes 16-31: row M=lane-16,  halves = K {8..15, 24..31}
__device__ __forceinline__ v16h load_frag_a(const h16* p, int ld) {
  int lane = threadIdx.x & 31;
  const h16* row = p + (size_t)(lane & 15) * ld;
  int kb = (lane < 16) ? 0 : 8;
  v16h a;
#pragma unroll
  for (int i = 0; i < 8; ++i) a[i] = row[kb + i];
#pragma unroll
  for (int i = 0; i < 8; ++i) a[8 + i] = row[kb + 16 + i];
  return a;
}

// B 32x16 f16, K-major source B[k][n] with leading dim `ld` (halves):
//   lanes 0-15 : col N=lane,     halves = K {0..7, 16..23}
//   lanes 16-31: col N=lane-16,  halves = K {8..15, 24..31}
__device__ __forceinline__ v16h load_frag_b(const h16* p, int ld) {
  int lane = threadIdx.x & 31;
  int col = lane & 15;
  int kb = (lane < 16) ? 0 : 8;
  v16h b;
#pragma unroll
  for (int i = 0; i < 16; ++i) b[i] = p[(size_t)(kb + i + ((i >> 3) << 3)) * ld + col];
  return b;
}

// ---------------------------------------------------------------------------
// Kernel 0: zero BN statistics region of the workspace
// ---------------------------------------------------------------------------
__global__ void zero_stats_kernel(float* wsf) {
  int i = threadIdx.x;
  wsf[i] = 0.f;
  wsf[i + 256] = 0.f;
}

// ---------------------------------------------------------------------------
// Kernel 1: conv1 (Cin=1, Cout=16, K=3) + ReLU, f16 store, BN1 stats.
// One block per batch row; thread (c = tid>>4, g = tid&15) owns a channel
// slice -> exactly one pair of LDS float atomics per thread.
// ---------------------------------------------------------------------------
__global__ void conv1_kernel(const float* __restrict__ x,
                             const float* __restrict__ w1,
                             const float* __restrict__ b1,
                             h16* __restrict__ act1,
                             float* __restrict__ stats1) {
  __shared__ float xs[SEQL];
  __shared__ float ws[48];
  __shared__ float bs[16];
  __shared__ float sstat[32];
  const int b = blockIdx.x, tid = threadIdx.x;
  if (tid < SEQL) xs[tid] = x[(size_t)b * SEQL + tid];
  if (tid < 48)  ws[tid] = w1[tid];
  if (tid < 16)  bs[tid] = b1[tid];
  if (tid < 32)  sstat[tid] = 0.f;
  __syncthreads();
  const int c = tid >> 4, g = tid & 15;
  float w0 = ws[c * 3], w1v = ws[c * 3 + 1], w2v = ws[c * 3 + 2], bb = bs[c];
  float su = 0.f, sq = 0.f;
  for (int t = g; t < 148; t += 16) {
    float y = w0 * xs[t] + w1v * xs[t + 1] + w2v * xs[t + 2] + bb;
    y = fmaxf(y, 0.f);
    act1[((size_t)b * 16 + c) * 148 + t] = (h16)y;
    su += y; sq += y * y;
  }
  atomicAdd(&sstat[c], su);
  atomicAdd(&sstat[16 + c], sq);
  __syncthreads();
  if (tid < 32) atomicAdd(&stats1[tid], sstat[tid]);
}

// ---------------------------------------------------------------------------
// BN finalize: scale = gamma*rsqrt(var+eps), shift = beta - mean*scale
// ---------------------------------------------------------------------------
__global__ void bn_finalize_kernel(const float* __restrict__ stats,
                                   const float* __restrict__ gamma,
                                   const float* __restrict__ beta,
                                   float* __restrict__ scale,
                                   float* __restrict__ shift,
                                   int nch, float inv_count) {
  int c = threadIdx.x;
  if (c < nch) {
    float mean = stats[c] * inv_count;
    float var = stats[nch + c] * inv_count - mean * mean;
    float sc = gamma[c] * rsqrtf(var + EPSBN);
    scale[c] = sc;
    shift[c] = beta[c] - mean * sc;
  }
}

// ---------------------------------------------------------------------------
// Kernel 2: BN1-apply + maxpool2 + conv2 (16->32ch, K=48) + ReLU + BN2 stats.
// One block (8 waves) per batch row. GEMM: M=72 rows(t), N=32, K=48 pad 64.
// ---------------------------------------------------------------------------
__global__ void conv2_kernel(const h16* __restrict__ act1,
                             const float* __restrict__ w2,
                             const float* __restrict__ b2,
                             const float* __restrict__ sc1,
                             const float* __restrict__ sh1,
                             h16* __restrict__ act2,
                             float* __restrict__ stats2) {
  __shared__ h16 Ax[80][64];      // M(pad 80) x K(pad 64)
  __shared__ h16 Bw[64][32];      // K x N
  __shared__ float x2s[16][76];   // pooled+BN'ed layer-1 activations
  __shared__ float sstat[64];
  __shared__ float bs2[32];
  const int b = blockIdx.x, tid = threadIdx.x;
  if (tid < 64) sstat[tid] = 0.f;
  if (tid < 32) bs2[tid] = b2[tid];
  // BN1 + maxpool2
  for (int i = tid; i < 16 * 74; i += 256) {
    int ci = i / 74, t = i - ci * 74;
    const h16* p = act1 + ((size_t)b * 16 + ci) * 148 + 2 * t;
    float s = sc1[ci], sh = sh1[ci];
    float a0 = s * (float)p[0] + sh;
    float a1 = s * (float)p[1] + sh;
    x2s[ci][t] = fmaxf(a0, a1);
  }
  // weights -> B tile (K-major), zero-pad K 48..63
  for (int i = tid; i < 64 * 32; i += 256) {
    int k = i >> 5, co = i & 31;
    float v = (k < 48) ? w2[co * 48 + k] : 0.f;
    Bw[k][co] = (h16)v;
  }
  __syncthreads();
  // im2col patches -> A tile
  for (int i = tid; i < 80 * 64; i += 256) {
    int row = i >> 6, k = i & 63;
    float v = 0.f;
    if (row < 72 && k < 48) { int ci = k / 3, kk = k - ci * 3; v = x2s[ci][row + kk]; }
    Ax[row][k] = (h16)v;
  }
  __syncthreads();
  const int wave = tid >> 5, lane = tid & 31;
  for (int tile = wave; tile < 10; tile += 8) {     // 5 M-tiles x 2 N-tiles
    int mt = tile >> 1, nt = tile & 1;
    v8f acc = {};
    v16h a0 = load_frag_a(&Ax[mt * 16][0], 64);
    v16h bf0 = load_frag_b(&Bw[0][nt * 16], 32);
    acc = __builtin_amdgcn_wmma_f32_16x16x32_f16(false, a0, false, bf0, (short)0, acc, false, false);
    v16h a1 = load_frag_a(&Ax[mt * 16][32], 64);
    v16h bf1 = load_frag_b(&Bw[32][nt * 16], 32);
    acc = __builtin_amdgcn_wmma_f32_16x16x32_f16(false, a1, false, bf1, (short)0, acc, false, false);
    int co = nt * 16 + (lane & 15);
    int rb = mt * 16 + ((lane < 16) ? 0 : 8);
    if (rb < 72) {                     // uniform over the 8 contiguous rows
      float bias = bs2[co];
      v8h pk;
      float su = 0.f, sq = 0.f;
#pragma unroll
      for (int r = 0; r < 8; ++r) {
        float y = fmaxf(acc[r] + bias, 0.f);
        pk[r] = (h16)y;
        su += y; sq += y * y;
      }
      *(v8h*)(act2 + ((size_t)b * 32 + co) * 72 + rb) = pk;  // 16B store
      atomicAdd(&sstat[co], su);
      atomicAdd(&sstat[32 + co], sq);
    }
  }
  __syncthreads();
  if (tid < 64) atomicAdd(&stats2[tid], sstat[tid]);
}

// ---------------------------------------------------------------------------
// Kernel 3: BN2-apply + maxpool2 + conv3 (32->64ch, K=96) + ReLU + BN3 stats.
// GEMM: M=34 rows (pad 48), N=64, K=96 (3 WMMA steps).
// act3 rows padded to 40 so lanes always store 8 contiguous halves; padded
// rows (t=34..39) are never read (avgpool8 touches t<32 only) and are
// excluded from BN3 statistics with a branchless select.
// ---------------------------------------------------------------------------
__global__ void conv3_kernel(const h16* __restrict__ act2,
                             const float* __restrict__ w3,
                             const float* __restrict__ b3,
                             const float* __restrict__ sc2,
                             const float* __restrict__ sh2,
                             h16* __restrict__ act3,
                             float* __restrict__ stats3) {
  __shared__ h16 Ax[48][96];
  __shared__ h16 Bw[96][64];
  __shared__ float x3s[32][40];
  __shared__ float sstat[128];
  __shared__ float bs3[64];
  const int b = blockIdx.x, tid = threadIdx.x;
  if (tid < 128) sstat[tid] = 0.f;
  if (tid < 64) bs3[tid] = b3[tid];
  for (int i = tid; i < 32 * 36; i += 256) {
    int ci = i / 36, t = i - ci * 36;
    const h16* p = act2 + ((size_t)b * 32 + ci) * 72 + 2 * t;
    float s = sc2[ci], sh = sh2[ci];
    float a0 = s * (float)p[0] + sh;
    float a1 = s * (float)p[1] + sh;
    x3s[ci][t] = fmaxf(a0, a1);
  }
  for (int i = tid; i < 96 * 64; i += 256) {
    int k = i >> 6, co = i & 63;
    Bw[k][co] = (h16)w3[co * 96 + k];
  }
  __syncthreads();
  for (int i = tid; i < 48 * 96; i += 256) {
    int row = i / 96, k = i - row * 96;
    float v = 0.f;
    if (row < 34) { int ci = k / 3, kk = k - ci * 3; v = x3s[ci][row + kk]; }
    Ax[row][k] = (h16)v;
  }
  __syncthreads();
  const int wave = tid >> 5, lane = tid & 31;
  for (int tile = wave; tile < 12; tile += 8) {     // 3 M-tiles x 4 N-tiles
    int mt = tile >> 2, nt = tile & 3;
    v8f acc = {};
#pragma unroll
    for (int ks = 0; ks < 3; ++ks) {
      v16h a = load_frag_a(&Ax[mt * 16][ks * 32], 96);
      v16h bf = load_frag_b(&Bw[ks * 32][nt * 16], 64);
      acc = __builtin_amdgcn_wmma_f32_16x16x32_f16(false, a, false, bf, (short)0, acc, false, false);
    }
    int co = nt * 16 + (lane & 15);
    int rb = mt * 16 + ((lane < 16) ? 0 : 8);
    if (rb < 40) {
      float bias = bs3[co];
      v8h pk;
      float su = 0.f, sq = 0.f;
#pragma unroll
      for (int r = 0; r < 8; ++r) {
        int t = rb + r;
        float y = fmaxf(acc[r] + bias, 0.f);
        pk[r] = (h16)y;
        float m = (t < 34) ? y : 0.f;   // exclude padded rows from stats
        su += m; sq += m * m;
      }
      *(v8h*)(act3 + ((size_t)b * 64 + co) * 40 + rb) = pk;  // 16B store
      atomicAdd(&sstat[co], su);
      atomicAdd(&sstat[64 + co], sq);
    }
  }
  __syncthreads();
  if (tid < 128) atomicAdd(&stats3[tid], sstat[tid]);
}

// ---------------------------------------------------------------------------
// Kernel 4: BN3-apply + avgpool8 (linear -> exact through pooling), then
// FC: v[B,150] = x[B,256] @ wfc[150,256]^T + bfc.
// Block = 64 batch rows x 160 cols (pad of 150); K=256 in 8 chunks of 32.
// Per wave: 5 persistent v8f accumulators (40 tiles / 8 waves).
// vout padded to 160 cols -> unconditional epilogue stores.
// ---------------------------------------------------------------------------
__global__ void fc_kernel(const h16* __restrict__ act3,
                          const float* __restrict__ wfc,
                          const float* __restrict__ bfc,
                          const float* __restrict__ sc3,
                          const float* __restrict__ sh3,
                          float* __restrict__ vout) {
  __shared__ h16 xfc[64][256];    // 32 KB: pooled features for 64 rows
  __shared__ h16 Bsub[32][160];   // 10 KB: one K-chunk of wfc^T
  const int b0 = blockIdx.x * 64, tid = threadIdx.x;
  // BN3 + avgpool8 -> flattened [64][256] (index = ci*4 + p); act3 stride 40
  for (int i = tid; i < 64 * 256; i += 256) {
    int m = i >> 8, k = i & 255;
    int ci = k >> 2, p = k & 3;
    v8h av = *(const v8h*)(act3 + ((size_t)(b0 + m) * 64 + ci) * 40 + p * 8);
    float s = 0.f;
#pragma unroll
    for (int j = 0; j < 8; ++j) s += (float)av[j];
    xfc[m][k] = (h16)(sc3[ci] * (s * 0.125f) + sh3[ci]);
  }
  v8f acc[5];
#pragma unroll
  for (int j = 0; j < 5; ++j) acc[j] = (v8f){};
  const int wave = tid >> 5, lane = tid & 31;
  for (int kc = 0; kc < 8; ++kc) {
    __syncthreads();
    if (kc < 7) __builtin_prefetch(&wfc[(kc + 1) * 32], 0, 3);  // global_prefetch_b8
    for (int i = tid; i < 32 * 160; i += 256) {
      int kk = i / 160, n = i - kk * 160;
      float v = (n < SEQL) ? wfc[(size_t)n * 256 + kc * 32 + kk] : 0.f;
      Bsub[kk][n] = (h16)v;
    }
    __syncthreads();
#pragma unroll
    for (int j = 0; j < 5; ++j) {
      int tt = wave + 8 * j;                 // 0..39
      int mt = tt / 10, nt = tt - mt * 10;   // 4 M-tiles x 10 N-tiles
      v16h a = load_frag_a(&xfc[mt * 16][kc * 32], 256);
      v16h bf = load_frag_b(&Bsub[0][nt * 16], 160);
      acc[j] = __builtin_amdgcn_wmma_f32_16x16x32_f16(false, a, false, bf, (short)0, acc[j], false, false);
    }
  }
#pragma unroll
  for (int j = 0; j < 5; ++j) {
    int tt = wave + 8 * j;
    int mt = tt / 10, nt = tt - mt * 10;
    int n = nt * 16 + (lane & 15);
    int rb = mt * 16 + ((lane < 16) ? 0 : 8);
    float bias = bfc[min(n, SEQL - 1)];      // branchless (cols >=150 unused)
#pragma unroll
    for (int r = 0; r < 8; ++r) {
      int m = rb + r;
      vout[(size_t)(b0 + m) * 160 + n] = acc[j][r] + bias;
    }
  }
}

// ---------------------------------------------------------------------------
// Kernel 5: per-row SRVF post-processing. One block per batch row; LDS
// Hillis-Steele scan implements the normalized cumulative trapezoid.
// ---------------------------------------------------------------------------
__device__ __forceinline__ void cumtrapz_norm_lds(const float* src, float* s,
                                                  float* dst, int i) {
  if (i < SEQL - 1) s[i] = 0.5f * (src[i] + src[i + 1]);
  __syncthreads();
  for (int off = 1; off < SEQL - 1; off <<= 1) {
    float t = 0.f;
    if (i < SEQL - 1 && i >= off) t = s[i - off];
    __syncthreads();
    if (i < SEQL - 1) s[i] += t;
    __syncthreads();
  }
  float tot = s[SEQL - 2];
  if (i < SEQL) dst[i] = (i == 0) ? 0.f : s[i - 1] / tot;
  __syncthreads();
}

__global__ void post_kernel(const float* __restrict__ vbuf,
                            const float* __restrict__ input,
                            float* __restrict__ out) {
  __shared__ float aa[152], ss[152], r1[152], rr[152], ff[152], qq[152];
  const int b = blockIdx.x, i = threadIdx.x;
  if (i < SEQL) {
    float vv = vbuf[(size_t)b * 160 + i];
    aa[i] = vv * vv;
    ff[i] = input[(size_t)b * SEQL + i];
  }
  __syncthreads();
  cumtrapz_norm_lds(aa, ss, r1, i);   // warp
  cumtrapz_norm_lds(r1, ss, rr, i);   // smooth integral
  const float la = (float)(SEQL - 1);
  if (i < SEQL) {
    float gf;
    if (i == 0)             gf = ff[1] - ff[0];
    else if (i == SEQL - 1) gf = ff[SEQL - 1] - ff[SEQL - 2];
    else                    gf = 0.5f * (ff[i + 1] - ff[i - 1]);
    gf *= la;
    float sg = (gf > 0.f) ? 1.f : ((gf < 0.f) ? -1.f : 0.f);
    qq[i] = sg * sqrtf(fabsf(gf));
  }
  __syncthreads();
  if (i < SEQL) {
    float gr;
    if (i == 0)             gr = rr[1] - rr[0];
    else if (i == SEQL - 1) gr = rr[SEQL - 1] - rr[SEQL - 2];
    else                    gr = 0.5f * (rr[i + 1] - rr[i - 1]);
    gr *= la;
    float xn = rr[i] * la;
    int idx = (int)floorf(xn);
    idx = min(max(idx, 0), SEQL - 2);
    float frac = xn - (float)idx;
    float q_r = qq[idx] + frac * (qq[idx + 1] - qq[idx]);
    float f_r = ff[idx] + frac * (ff[idx + 1] - ff[idx]);
    size_t o = (size_t)b * SEQL + i;
    out[o] = q_r * sqrtf(gr);
    out[(size_t)BATCH * SEQL + o] = f_r;
    out[(size_t)2 * BATCH * SEQL + o] = rr[i];
  }
}

// ---------------------------------------------------------------------------
// Host-side launch
// ---------------------------------------------------------------------------
extern "C" void kernel_launch(void* const* d_in, const int* in_sizes, int n_in,
                              void* d_out, int out_size, void* d_ws, size_t ws_size,
                              hipStream_t stream) {
  (void)in_sizes; (void)n_in; (void)out_size; (void)ws_size;
  const float* input = (const float*)d_in[0];
  const float* w1 = (const float*)d_in[1];
  const float* b1 = (const float*)d_in[2];
  const float* g1 = (const float*)d_in[3];
  const float* be1 = (const float*)d_in[4];
  const float* w2 = (const float*)d_in[5];
  const float* b2 = (const float*)d_in[6];
  const float* g2 = (const float*)d_in[7];
  const float* be2 = (const float*)d_in[8];
  const float* w3 = (const float*)d_in[9];
  const float* b3 = (const float*)d_in[10];
  const float* g3 = (const float*)d_in[11];
  const float* be3 = (const float*)d_in[12];
  const float* wfc = (const float*)d_in[13];
  const float* bfc = (const float*)d_in[14];
  float* out = (float*)d_out;

  // workspace layout
  float* wsf = (float*)d_ws;
  float* stats1 = wsf;            // 32 floats  (sum16, sumsq16)
  float* stats2 = wsf + 32;       // 64 floats
  float* stats3 = wsf + 96;       // 128 floats
  float* sc1 = wsf + 256, *sh1 = wsf + 272;
  float* sc2 = wsf + 288, *sh2 = wsf + 320;
  float* sc3 = wsf + 352, *sh3 = wsf + 416;
  char* big = (char*)d_ws + 4096;
  h16* act1 = (h16*)big;                                    // B*16*148 halves
  h16* act2 = act1 + (size_t)BATCH * 16 * 148;              // B*32*72
  h16* act3 = act2 + (size_t)BATCH * 32 * 72;               // B*64*40 (padded)
  float* vbuf = (float*)(act3 + (size_t)BATCH * 64 * 40);   // B*160 floats

  zero_stats_kernel<<<1, 256, 0, stream>>>(wsf);

  conv1_kernel<<<BATCH, 256, 0, stream>>>(input, w1, b1, act1, stats1);
  bn_finalize_kernel<<<1, 64, 0, stream>>>(stats1, g1, be1, sc1, sh1, 16,
                                           1.0f / ((float)BATCH * 148.f));

  conv2_kernel<<<BATCH, 256, 0, stream>>>(act1, w2, b2, sc1, sh1, act2, stats2);
  bn_finalize_kernel<<<1, 64, 0, stream>>>(stats2, g2, be2, sc2, sh2, 32,
                                           1.0f / ((float)BATCH * 72.f));

  conv3_kernel<<<BATCH, 256, 0, stream>>>(act2, w3, b3, sc2, sh2, act3, stats3);
  bn_finalize_kernel<<<1, 64, 0, stream>>>(stats3, g3, be3, sc3, sh3, 64,
                                           1.0f / ((float)BATCH * 34.f));

  fc_kernel<<<BATCH / 64, 256, 0, stream>>>(act3, wfc, bfc, sc3, sh3, vbuf);

  post_kernel<<<BATCH, 256, 0, stream>>>(vbuf, input, out);
}
